// Model_39676907886721
// MI455X (gfx1250) — matrix operand order repaired and checked by reading, not compiled, not run on hardware
//
#include <hip/hip_runtime.h>
#include <hip/hip_bf16.h>

typedef __bf16 bf16_t;
typedef __attribute__((ext_vector_type(16))) __bf16 v16bf;
typedef __attribute__((ext_vector_type(8)))  float  v8f;
typedef __attribute__((ext_vector_type(4)))  int    v4i_b;

#define DIM  1024
#define SEQ  2048
#define NB   4
#define SOFTMAX_SCALE (1.0f / 32.0f)

// ---------------------------------------------------------------------------
// CDNA5 async global->LDS copy (ASYNCcnt-tracked, no VGPR round trip).
// clang-22 signature: (as1 v4i* global_src, as3 v4i* lds_dst, imm off, imm cpol)
// Guarded with __has_builtin so an older toolchain falls back to manual copy.
// ---------------------------------------------------------------------------
#if __has_builtin(__builtin_amdgcn_global_load_async_to_lds_b128)
#define HAVE_ASYNC_COPY 1
typedef __attribute__((address_space(1))) v4i_b* gas_v4i;
typedef __attribute__((address_space(3))) v4i_b* las_v4i;

__device__ __forceinline__ void async_copy_b128(const void* g, void* l) {
  __builtin_amdgcn_global_load_async_to_lds_b128(
      (gas_v4i)g, (las_v4i)l, /*imm offset=*/0, /*cpol=*/0);
}
__device__ __forceinline__ void wait_async0() {
#if __has_builtin(__builtin_amdgcn_s_wait_asynccnt)
  __builtin_amdgcn_s_wait_asynccnt(0);
#else
  asm volatile("s_wait_asynccnt 0" ::: "memory");
#endif
}
#endif

// ---------------------------------------------------------------------------
// Fragment loaders per CDNA5 WMMA VGPR layouts (ISA 7.12.2, 16-bit operands).
// A 16x32: lane m = lane%16, element 2v+pos holds K = (v%4)*2 + (v/4)*16 + 8*half + pos
// B 32x16: lane n = lane%16, element 2v+pos holds K = (v%4)*2 + (v/4)*16 + 8*half + pos
// C 16x16 f32: VGPR r holds M = r + 8*half, N = lane%16
// ---------------------------------------------------------------------------
__device__ __forceinline__ v16bf load_a_frag(const bf16_t* __restrict__ row, int half) {
  v16bf a;
#pragma unroll
  for (int v = 0; v < 8; ++v) {
    const int k = ((v & 3) << 1) + ((v >> 2) << 4) + (half << 3);
    a[2 * v]     = row[k];
    a[2 * v + 1] = row[k + 1];
  }
  return a;
}

__device__ __forceinline__ v16bf load_b_frag(const bf16_t* __restrict__ col, int half) {
  v16bf b;
#pragma unroll
  for (int v = 0; v < 8; ++v) {
    const int k = ((v & 3) << 1) + ((v >> 2) << 4) + (half << 3);
    b[2 * v]     = col[k];
    b[2 * v + 1] = col[k + 1];
  }
  return b;
}

__device__ __forceinline__ v8f wmma_bf16(v16bf a, v16bf b, v8f c) {
  return __builtin_amdgcn_wmma_f32_16x16x32_bf16(
      /*neg_a=*/false, a, /*neg_b=*/false, b,
      /*c_mod=*/(short)0, c, /*reuse_a=*/false, /*reuse_b=*/false);
}

// ---------------------------------------------------------------------------
// Kernel 1: fused QKV projection.  y = x @ W^T + b, output bf16.
// Block tile 128(M) x 128(N); 8 waves, wave tile 32x64 (2x4 WMMA accums).
// z = 0 -> Q, z = 1 -> K (row-major [m][n]); z = 2 -> V stored transposed
// as vT[b][d][s] so attention B-fragments read contiguous pairs along keys.
// Staging converts f32 -> bf16, so it stays a manual VGPR copy.
// ---------------------------------------------------------------------------
__global__ __launch_bounds__(256) void qkv_proj_kernel(
    const float* __restrict__ x,
    const float* __restrict__ Wq, const float* __restrict__ bq,
    const float* __restrict__ Wk, const float* __restrict__ bk,
    const float* __restrict__ Wv, const float* __restrict__ bv,
    bf16_t* __restrict__ qo, bf16_t* __restrict__ ko, bf16_t* __restrict__ vto) {
  __shared__ alignas(16) bf16_t Xs[128][32];
  __shared__ alignas(16) bf16_t Ws[128][32];   // Ws[n][k] = W[n0+n][kc+k]

  const int z    = blockIdx.z;
  const float* W  = (z == 0) ? Wq : ((z == 1) ? Wk : Wv);
  const float* bi = (z == 0) ? bq : ((z == 1) ? bk : bv);

  const int tid  = threadIdx.x;
  const int w    = tid >> 5;
  const int lane = tid & 31;
  const int half = lane >> 4;
  const int ln   = lane & 15;
  const int mw   = w >> 1;          // 0..3  (32-row slice)
  const int nw   = w & 1;           // 0..1  (64-col slice)
  const int m0   = blockIdx.x * 128;
  const int n0   = blockIdx.y * 128;

  const v8f vzero = {0.f, 0.f, 0.f, 0.f, 0.f, 0.f, 0.f, 0.f};
  v8f acc[2][4];
#pragma unroll
  for (int mt = 0; mt < 2; ++mt)
#pragma unroll
    for (int nt = 0; nt < 4; ++nt) acc[mt][nt] = vzero;

  for (int kc = 0; kc < DIM; kc += 32) {
    __syncthreads();
#pragma unroll
    for (int i = tid; i < 128 * 32; i += 256) {
      const int r = i >> 5, c = i & 31;
      Xs[r][c] = (bf16_t)x[(size_t)(m0 + r) * DIM + kc + c];
      Ws[r][c] = (bf16_t)W[(size_t)(n0 + r) * DIM + kc + c];
    }
    if (kc + 32 < DIM)  // speculative prefetch of next x tile (global_prefetch_b8)
      __builtin_prefetch(&x[(size_t)(m0 + (tid >> 1)) * DIM + kc + 32], 0, 0);
    __syncthreads();

    v16bf af[2], bf[4];
#pragma unroll
    for (int mt = 0; mt < 2; ++mt)
      af[mt] = load_a_frag(&Xs[mw * 32 + mt * 16 + ln][0], half);
#pragma unroll
    for (int nt = 0; nt < 4; ++nt)
      bf[nt] = load_b_frag(&Ws[nw * 64 + nt * 16 + ln][0], half);
#pragma unroll
    for (int mt = 0; mt < 2; ++mt)
#pragma unroll
      for (int nt = 0; nt < 4; ++nt)
        acc[mt][nt] = wmma_bf16(af[mt], bf[nt], acc[mt][nt]);
  }

  // Epilogue: bias add, convert to bf16, store.
#pragma unroll
  for (int mt = 0; mt < 2; ++mt) {
#pragma unroll
    for (int nt = 0; nt < 4; ++nt) {
      const int n_g = n0 + nw * 64 + nt * 16 + ln;
      const float bias_v = bi[n_g];
#pragma unroll
      for (int r = 0; r < 8; ++r) {
        const int m_g = m0 + mw * 32 + mt * 16 + r + half * 8;
        const float val = acc[mt][nt][r] + bias_v;
        if (z == 2) {
          const int bb = m_g >> 11;          // m / SEQ
          const int ss = m_g & (SEQ - 1);    // m % SEQ
          vto[(size_t)bb * DIM * SEQ + (size_t)n_g * SEQ + ss] = (bf16_t)val;
        } else {
          bf16_t* o = (z == 0) ? qo : ko;
          o[(size_t)m_g * DIM + n_g] = (bf16_t)val;
        }
      }
    }
  }
}

// ---------------------------------------------------------------------------
// Kernel 2: flash attention.  Block = (batch b, 32 query rows).
// Key blocks of 64.  Phase 1: 8 waves compute 32x64 score tile (one 16x16
// WMMA tile per wave, D staged 64-wide in LDS via async global->LDS copies).
// Phase 2: online softmax (wave 0).  Phase 3: each wave owns a 128-wide
// d-slice of the O accumulator (2x8 WMMA tiles); P from LDS, V^T from global.
// ---------------------------------------------------------------------------
__global__ __launch_bounds__(256) void flash_attn_kernel(
    const bf16_t* __restrict__ q, const bf16_t* __restrict__ k,
    const bf16_t* __restrict__ vt, float* __restrict__ out) {
  __shared__ alignas(16) bf16_t Qs[32][64];
  __shared__ alignas(16) bf16_t Ks[64][64];
  __shared__ float  Sc[32][64];
  __shared__ alignas(16) bf16_t Ps[32][64];
  __shared__ float  m_s[32], l_s[32], a_s[32];

  const int tid  = threadIdx.x;
  const int w    = tid >> 5;
  const int lane = tid & 31;
  const int half = lane >> 4;
  const int ln   = lane & 15;
  const int b    = blockIdx.y;
  const int q0   = blockIdx.x * 32;
  const int mt1  = w >> 2;          // 0..1
  const int nt1  = w & 3;           // 0..3
  const size_t qrow0 = (size_t)(b * SEQ + q0);
  const size_t krow0 = (size_t)(b * SEQ);
  const size_t vbase = (size_t)b * DIM * SEQ;

  if (tid < 32) { m_s[tid] = -1e30f; l_s[tid] = 0.f; }

  const v8f vzero = {0.f, 0.f, 0.f, 0.f, 0.f, 0.f, 0.f, 0.f};
  v8f acc_o[2][8];
#pragma unroll
  for (int mt = 0; mt < 2; ++mt)
#pragma unroll
    for (int nt = 0; nt < 8; ++nt) acc_o[mt][nt] = vzero;

  for (int kb = 0; kb < SEQ; kb += 64) {
    // ---- Phase 1: scores S = Q @ K^T for this 64-key block ----
    v8f sc = vzero;
    for (int dc = 0; dc < DIM; dc += 64) {
      __syncthreads();
#ifdef HAVE_ASYNC_COPY
      {
        // Qs: 32x64 bf16 = 4 KB = 256 threads x one 16 B chunk.
        const int eq = tid * 8;                  // first bf16 element
        const int rq = eq >> 6, cq = eq & 63;
        async_copy_b128(&q[(qrow0 + rq) * DIM + dc + cq], &Qs[rq][cq]);
        // Ks: 64x64 bf16 = 8 KB = 256 threads x two 16 B chunks.
#pragma unroll
        for (int j = 0; j < 2; ++j) {
          const int ek = (tid + j * 256) * 8;
          const int rk = ek >> 6, ck = ek & 63;
          async_copy_b128(&k[(krow0 + kb + rk) * DIM + dc + ck], &Ks[rk][ck]);
        }
        wait_async0();
      }
#else
#pragma unroll
      for (int i = tid; i < 32 * 64; i += 256) {
        const int r = i >> 6, c = i & 63;
        Qs[r][c] = q[(qrow0 + r) * DIM + dc + c];
      }
#pragma unroll
      for (int i = tid; i < 64 * 64; i += 256) {
        const int r = i >> 6, c = i & 63;
        Ks[r][c] = k[(krow0 + kb + r) * DIM + dc + c];
      }
#endif
      __syncthreads();
#pragma unroll
      for (int kk = 0; kk < 64; kk += 32) {
        const v16bf af = load_a_frag(&Qs[mt1 * 16 + ln][kk], half);
        const v16bf bf = load_b_frag(&Ks[nt1 * 16 + ln][kk], half);
        sc = wmma_bf16(af, bf, sc);
      }
    }
#pragma unroll
    for (int r = 0; r < 8; ++r)
      Sc[mt1 * 16 + r + half * 8][nt1 * 16 + ln] = sc[r];
    __syncthreads();

    // ---- Phase 2: online softmax statistics (wave 0, lane == row) ----
    if (tid < 32) {
      const float mold = m_s[tid];
      float mb = -1e30f;
      for (int j = 0; j < 64; ++j) mb = fmaxf(mb, Sc[tid][j] * SOFTMAX_SCALE);
      const float mnew  = fmaxf(mold, mb);
      const float alpha = __expf(mold - mnew);
      float sum = 0.f;
      for (int j = 0; j < 64; ++j) {
        const float p = __expf(Sc[tid][j] * SOFTMAX_SCALE - mnew);
        Ps[tid][j] = (bf16_t)p;
        sum += p;
      }
      m_s[tid] = mnew;
      l_s[tid] = l_s[tid] * alpha + sum;
      a_s[tid] = alpha;
    }
    __syncthreads();

    // ---- Phase 3: O = O * alpha + P @ V  (wave owns d-slice [w*128, +128)) ----
#pragma unroll
    for (int mt = 0; mt < 2; ++mt)
#pragma unroll
      for (int r = 0; r < 8; ++r) {
        const float al = a_s[mt * 16 + r + half * 8];
#pragma unroll
        for (int nt = 0; nt < 8; ++nt) acc_o[mt][nt][r] *= al;
      }

    const int d0 = w * 128;
#pragma unroll
    for (int kk = 0; kk < 64; kk += 32) {
      v16bf pa[2];
      pa[0] = load_a_frag(&Ps[ln][kk], half);
      pa[1] = load_a_frag(&Ps[16 + ln][kk], half);
#pragma unroll
      for (int nt = 0; nt < 8; ++nt) {
        const int dg = d0 + nt * 16 + ln;
        const v16bf vb = load_b_frag(&vt[vbase + (size_t)dg * SEQ + kb + kk], half);
        acc_o[0][nt] = wmma_bf16(pa[0], vb, acc_o[0][nt]);
        acc_o[1][nt] = wmma_bf16(pa[1], vb, acc_o[1][nt]);
      }
    }
  }
  __syncthreads();

  // ---- Epilogue: out = O / l ----
  const int d0 = w * 128;
#pragma unroll
  for (int mt = 0; mt < 2; ++mt)
#pragma unroll
    for (int r = 0; r < 8; ++r) {
      const int m = mt * 16 + r + half * 8;
      const float inv_l = 1.0f / l_s[m];
#pragma unroll
      for (int nt = 0; nt < 8; ++nt)
        out[(qrow0 + m) * DIM + d0 + nt * 16 + ln] = acc_o[mt][nt][r] * inv_l;
    }
}

// ---------------------------------------------------------------------------
// Host-side launch.
// ---------------------------------------------------------------------------
extern "C" void kernel_launch(void* const* d_in, const int* in_sizes, int n_in,
                              void* d_out, int out_size, void* d_ws, size_t ws_size,
                              hipStream_t stream) {
  (void)in_sizes; (void)n_in; (void)out_size; (void)ws_size;

  const float* x1 = (const float*)d_in[0];
  const float* Wq = (const float*)d_in[1];
  const float* bq = (const float*)d_in[2];
  const float* Wk = (const float*)d_in[3];
  const float* bk = (const float*)d_in[4];
  const float* Wv = (const float*)d_in[5];
  const float* bv = (const float*)d_in[6];

  const size_t MN = (size_t)NB * SEQ * DIM;   // 8192*1024 elements
  bf16_t* qws  = (bf16_t*)d_ws;
  bf16_t* kws  = qws + MN;
  bf16_t* vtws = kws + MN;                    // V stored transposed [b][d][s]

  dim3 gq(NB * SEQ / 128, DIM / 128, 3);      // (64, 8, 3)
  qkv_proj_kernel<<<gq, 256, 0, stream>>>(x1, Wq, bq, Wk, bk, Wv, bv,
                                          qws, kws, vtws);

  dim3 ga(SEQ / 32, NB);                      // (64, 4)
  flash_attn_kernel<<<ga, 256, 0, stream>>>(qws, kws, vtws, (float*)d_out);
}
